// MultiHeadAttention_82738249990377
// MI455X (gfx1250) — matrix operand order repaired and checked by reading, not compiled, not run on hardware
//
#include <hip/hip_runtime.h>
#include <hip/hip_bf16.h>

#define S_LEN 4096
#define EMB   512
#define HD    64
#define NH    8

typedef __attribute__((ext_vector_type(8)))  float  v8f;
typedef __attribute__((ext_vector_type(8)))  __bf16 v8bf;
typedef __attribute__((ext_vector_type(16))) __bf16 v16bf;

// ---------------------------------------------------------------------------
// WMMA fragment helpers (wave32, v_wmma_f32_16x16x32_bf16)
//
// A (16x32, MxK):  lane l: row m = l&15, half hh = l>>4.
//   elements 0..15 = K = 16*hh + 0..15
//   -> one contiguous 32B run from a row-major row.
// B (32x16, KxN):  lane l: col n = l&15, half hh = l>>4.
//   elements 0..15 = K = 16*hh + 0..15
//   -> one contiguous 32B run from the "transposed operand" row.
// C/D (16x16 f32): lane l<16: col l,    rows 0..7  (VGPR v -> M=v)
//                  lane l>=16: col l-16, rows 8..15 (VGPR v -> M=8+v)
// ---------------------------------------------------------------------------

static __device__ inline v16bf frag_a(const __bf16* rowptr, int kk, int hh) {
    const __bf16* p = rowptr + kk + hh * 16;
    v8bf lo = *(const v8bf*)p;
    v8bf hi = *(const v8bf*)(p + 8);
    v16bf r;
#pragma unroll
    for (int i = 0; i < 8; ++i) { r[i] = lo[i]; r[i + 8] = hi[i]; }
    return r;
}

static __device__ inline v16bf frag_b(const __bf16* p) {
    v8bf lo = *(const v8bf*)p;
    v8bf hi = *(const v8bf*)(p + 8);
    v16bf r;
#pragma unroll
    for (int i = 0; i < 8; ++i) { r[i] = lo[i]; r[i + 8] = hi[i]; }
    return r;
}

static __device__ inline v8f wmma_bf16(v16bf a, v16bf b, v8f c) {
    return __builtin_amdgcn_wmma_f32_16x16x32_bf16(
        /*neg_a=*/false, a, /*neg_b=*/false, b,
        /*c_mod=*/(short)0, c, /*reuse_a=*/false, /*reuse_b=*/false);
}

// ---------------------------------------------------------------------------
// Kernel 0: fp32 -> bf16 convert of X (8 elements / thread, vectorized)
// ---------------------------------------------------------------------------
__global__ void mha_cvt_kernel(const float* __restrict__ X,
                               __bf16* __restrict__ Xb) {
    int i = (blockIdx.x * blockDim.x + threadIdx.x) * 8;
    float4 f0 = *(const float4*)(X + i);
    float4 f1 = *(const float4*)(X + i + 4);
    v8bf pk;
    pk[0] = (__bf16)f0.x; pk[1] = (__bf16)f0.y;
    pk[2] = (__bf16)f0.z; pk[3] = (__bf16)f0.w;
    pk[4] = (__bf16)f1.x; pk[5] = (__bf16)f1.y;
    pk[6] = (__bf16)f1.z; pk[7] = (__bf16)f1.w;
    *(v8bf*)(Xb + i) = pk;
}

// ---------------------------------------------------------------------------
// Kernel 1: QKV projection.  Q,K -> [H][S][D] bf16 ; V -> transposed [H][D][S]
// Q is pre-scaled by 1/sqrt(D) so attention needs no score scaling.
// grid: H * (S/256) blocks, 256 threads (8 waves, 32 rows each -> each LDS
// B-fragment feeds two WMMAs).  kk loop unroll capped at 2 to avoid spills.
// dynamic LDS: transposed weight tile Wt[64][EMB+8] bf16
// ---------------------------------------------------------------------------
__global__ void mha_proj_kernel(const __bf16* __restrict__ Xb,
                                const float* __restrict__ Wq,
                                const float* __restrict__ Wk,
                                const float* __restrict__ Wv,
                                __bf16* __restrict__ Q,
                                __bf16* __restrict__ K,
                                __bf16* __restrict__ Vt) {
    extern __shared__ __bf16 lds[];
    const int STRIDE = EMB + 8;

    const int wave = threadIdx.x >> 5;
    const int lane = threadIdx.x & 31;
    const int hh = lane >> 4, nn = lane & 15, mbase = hh * 8;

    const int h    = blockIdx.x / (S_LEN / 256);
    const int rblk = blockIdx.x % (S_LEN / 256);
    const int s0   = rblk * 256 + wave * 32;      // this wave: rows s0 .. s0+31

    const __bf16* xrow0 = Xb + (size_t)(s0 + nn) * EMB;
    const __bf16* xrow1 = Xb + (size_t)(s0 + 16 + nn) * EMB;

#pragma unroll 1
    for (int w = 0; w < 3; ++w) {
        // branchless source select (no runtime-indexed stack array)
        const float* Wsrc = (w == 0) ? Wq : ((w == 1) ? Wk : Wv);
        Wsrc += (size_t)h * EMB * HD;

        // cooperative load of W[h] (fp32 [E][D]) into LDS, transposed to bf16 [D][E]
        for (int idx = threadIdx.x; idx < EMB * HD; idx += 256) {
            int e = idx >> 6;       // E index
            int d = idx & 63;       // D index
            lds[d * STRIDE + e] = (__bf16)Wsrc[idx];
        }
        __syncthreads();

        v8f za = {0,0,0,0,0,0,0,0};
        v8f accA[4] = {za, za, za, za};           // rows s0..s0+15
        v8f accB[4] = {za, za, za, za};           // rows s0+16..s0+31
#pragma unroll 2
        for (int kk = 0; kk < EMB; kk += 32) {
            v16bf a0 = frag_a(xrow0, kk, hh);
            v16bf a1 = frag_a(xrow1, kk, hh);
#pragma unroll
            for (int j = 0; j < 4; ++j) {
                v16bf b = frag_b(lds + (j * 16 + nn) * STRIDE + kk + hh * 16);
                accA[j] = wmma_bf16(a0, b, accA[j]);
                accB[j] = wmma_bf16(a1, b, accB[j]);
            }
        }

        const float qscale = (w == 0) ? 0.125f : 1.0f;   // 1/sqrt(64) folded into Q
        if (w == 2) {
            // V stored transposed: Vt[h][d][s]; per lane col n fixed -> rows (s)
            // contiguous: one 16B packed store per tile.
#pragma unroll
            for (int j = 0; j < 4; ++j) {
                v8bf pkA, pkB;
#pragma unroll
                for (int v = 0; v < 8; ++v) {
                    pkA[v] = (__bf16)accA[j][v];
                    pkB[v] = (__bf16)accB[j][v];
                }
                __bf16* col = Vt + ((size_t)h * HD + j * 16 + nn) * S_LEN + s0 + mbase;
                *(v8bf*)(col)      = pkA;
                *(v8bf*)(col + 16) = pkB;
            }
        } else {
            __bf16* dst = (w == 0 ? Q : K) + (size_t)h * S_LEN * HD;
#pragma unroll
            for (int j = 0; j < 4; ++j)
#pragma unroll
                for (int v = 0; v < 8; ++v) {
                    dst[(size_t)(s0 + mbase + v) * HD + j * 16 + nn]      = (__bf16)(accA[j][v] * qscale);
                    dst[(size_t)(s0 + 16 + mbase + v) * HD + j * 16 + nn] = (__bf16)(accB[j][v] * qscale);
                }
        }
        __syncthreads();
    }
}

// ---------------------------------------------------------------------------
// Kernel 2: flash attention.  One wave owns 16 query rows of one head and
// streams all 4096 keys in chunks of 64:
//   8 score WMMAs + online softmax (row-max shfl-reduced over 16-lane halves,
//   row-sum kept as per-lane partial, reduced once in epilogue) + 8 PV WMMAs.
// P is re-laid-out C->A through a private per-wave LDS buffer (in-wave DS
// ordering guarantees RAW; only a compiler barrier is needed).
// grid: H * (S/64) blocks, 128 threads (4 waves)
// ---------------------------------------------------------------------------
__global__ void mha_attn_kernel(const __bf16* __restrict__ Q,
                                const __bf16* __restrict__ K,
                                const __bf16* __restrict__ Vt,
                                __bf16* __restrict__ Oc) {
    __shared__ __bf16 pbuf_all[4 * 16 * 64];   // 2KB per wave
    const int wave = threadIdx.x >> 5;
    const int lane = threadIdx.x & 31;
    const int hh = lane >> 4, nn = lane & 15, mbase = hh * 8;

    const int h    = blockIdx.x / (S_LEN / 64);
    const int qblk = blockIdx.x % (S_LEN / 64);
    const int s0   = qblk * 64 + wave * 16;

    __bf16* pbuf = pbuf_all + wave * (16 * 64);
    const __bf16* Qh = Q  + (size_t)h * S_LEN * HD;
    const __bf16* Kh = K  + (size_t)h * S_LEN * HD;
    const __bf16* Vh = Vt + (size_t)h * HD * S_LEN;

    const __bf16* qrow = Qh + (size_t)(s0 + nn) * HD;
    v16bf qa0 = frag_a(qrow, 0, hh);
    v16bf qa1 = frag_a(qrow, 32, hh);

    v8f zo = {0,0,0,0,0,0,0,0};
    v8f o0 = zo, o1 = zo, o2 = zo, o3 = zo;
    float mrow[8], lsum[8];
#pragma unroll
    for (int v = 0; v < 8; ++v) { mrow[v] = -1e30f; lsum[v] = 0.0f; }

#pragma unroll 1
    for (int k0 = 0; k0 < S_LEN; k0 += 64) {
        if (k0 + 64 < S_LEN) {
            __builtin_prefetch(Kh + (size_t)(k0 + 64 + nn * 4) * HD, 0, 0);
            __builtin_prefetch(Vh + (size_t)(nn * 4) * S_LEN + k0 + 64, 0, 0);
        }
        // scores: S = Q(16x64) . K^T(64x64), four 16-key tiles, K-dim chained 2x32
        v8f st[4] = {zo, zo, zo, zo};
#pragma unroll
        for (int t = 0; t < 4; ++t) {
            const __bf16* kb = Kh + (size_t)(k0 + t * 16 + nn) * HD;
            st[t] = wmma_bf16(qa0, frag_b(kb + 0  + hh * 16), st[t]);
            st[t] = wmma_bf16(qa1, frag_b(kb + 32 + hh * 16), st[t]);
        }

        // online softmax over 64 new columns; rows live across 16-lane halves
#pragma unroll
        for (int v = 0; v < 8; ++v) {
            float a0 = st[0][v], a1 = st[1][v], a2 = st[2][v], a3 = st[3][v];
            float mx = fmaxf(fmaxf(a0, a1), fmaxf(a2, a3));
            mx = fmaxf(mx, __shfl_xor(mx, 1));
            mx = fmaxf(mx, __shfl_xor(mx, 2));
            mx = fmaxf(mx, __shfl_xor(mx, 4));
            mx = fmaxf(mx, __shfl_xor(mx, 8));
            float nm = fmaxf(mrow[v], mx);
            float sc = __expf(mrow[v] - nm);
            float e0 = __expf(a0 - nm);
            float e1 = __expf(a1 - nm);
            float e2 = __expf(a2 - nm);
            float e3 = __expf(a3 - nm);
            lsum[v] = lsum[v] * sc + ((e0 + e1) + (e2 + e3));  // per-lane partial
            mrow[v] = nm;
            o0[v] *= sc; o1[v] *= sc; o2[v] *= sc; o3[v] *= sc;
            __bf16* pr = pbuf + (mbase + v) * 64 + nn;
            pr[0]  = (__bf16)e0;
            pr[16] = (__bf16)e1;
            pr[32] = (__bf16)e2;
            pr[48] = (__bf16)e3;
        }
        // DS ops are in-order within a wave -> only a compiler barrier needed
        asm volatile("" ::: "memory");

        v16bf pa0 = frag_a(pbuf + nn * 64, 0, hh);
        v16bf pa1 = frag_a(pbuf + nn * 64, 32, hh);
        {
            const __bf16* vb0 = Vh + (size_t)(0 * 16 + nn) * S_LEN + k0;
            const __bf16* vb1 = Vh + (size_t)(1 * 16 + nn) * S_LEN + k0;
            const __bf16* vb2 = Vh + (size_t)(2 * 16 + nn) * S_LEN + k0;
            const __bf16* vb3 = Vh + (size_t)(3 * 16 + nn) * S_LEN + k0;
            o0 = wmma_bf16(pa0, frag_b(vb0 + hh * 16), o0);
            o1 = wmma_bf16(pa0, frag_b(vb1 + hh * 16), o1);
            o2 = wmma_bf16(pa0, frag_b(vb2 + hh * 16), o2);
            o3 = wmma_bf16(pa0, frag_b(vb3 + hh * 16), o3);
            o0 = wmma_bf16(pa1, frag_b(vb0 + 32 + hh * 16), o0);
            o1 = wmma_bf16(pa1, frag_b(vb1 + 32 + hh * 16), o1);
            o2 = wmma_bf16(pa1, frag_b(vb2 + 32 + hh * 16), o2);
            o3 = wmma_bf16(pa1, frag_b(vb3 + 32 + hh * 16), o3);
        }
        asm volatile("" ::: "memory");   // WAR fence before next chunk's stores
    }

    // epilogue: reduce deferred row-sums, normalize, write Oc[S][H*D] (bf16)
    v8f oo[4] = {o0, o1, o2, o3};
#pragma unroll
    for (int v = 0; v < 8; ++v) {
        float l = lsum[v];
        l += __shfl_xor(l, 1);
        l += __shfl_xor(l, 2);
        l += __shfl_xor(l, 4);
        l += __shfl_xor(l, 8);
        float inv = 1.0f / l;
        size_t row = (size_t)(s0 + mbase + v) * EMB + h * HD;
#pragma unroll
        for (int j = 0; j < 4; ++j)
            Oc[row + j * 16 + nn] = (__bf16)(oo[j][v] * inv);
    }
}

// ---------------------------------------------------------------------------
// Kernel 3: output projection  Z[S][64] = Oc[S][512] @ W0[512][64]  (fp32 out)
// grid: S/128 blocks, 256 threads; dynamic LDS: W0 transposed bf16
// ---------------------------------------------------------------------------
__global__ void mha_outproj_kernel(const __bf16* __restrict__ Oc,
                                   const float* __restrict__ W0,
                                   float* __restrict__ Z) {
    extern __shared__ __bf16 lds[];
    const int STRIDE = EMB + 8;
    const int wave = threadIdx.x >> 5;
    const int lane = threadIdx.x & 31;
    const int hh = lane >> 4, nn = lane & 15, mbase = hh * 8;
    const int s0 = blockIdx.x * 128 + wave * 16;

    for (int idx = threadIdx.x; idx < EMB * HD; idx += 256) {
        int e = idx >> 6, d = idx & 63;
        lds[d * STRIDE + e] = (__bf16)W0[idx];
    }
    __syncthreads();

    const __bf16* orow = Oc + (size_t)(s0 + nn) * EMB;
    v8f za = {0,0,0,0,0,0,0,0};
    v8f acc[4] = {za, za, za, za};
#pragma unroll 4
    for (int kk = 0; kk < EMB; kk += 32) {
        v16bf a = frag_a(orow, kk, hh);
#pragma unroll
        for (int j = 0; j < 4; ++j)
            acc[j] = wmma_bf16(a, frag_b(lds + (j * 16 + nn) * STRIDE + kk + hh * 16), acc[j]);
    }
#pragma unroll
    for (int j = 0; j < 4; ++j)
#pragma unroll
        for (int v = 0; v < 8; ++v)
            Z[(size_t)(s0 + mbase + v) * HD + j * 16 + nn] = acc[j][v];
}

// ---------------------------------------------------------------------------
extern "C" void kernel_launch(void* const* d_in, const int* in_sizes, int n_in,
                              void* d_out, int out_size, void* d_ws, size_t ws_size,
                              hipStream_t stream) {
    const float* X  = (const float*)d_in[0];
    const float* Wq = (const float*)d_in[1];
    const float* Wk = (const float*)d_in[2];
    const float* Wv = (const float*)d_in[3];
    const float* W0 = (const float*)d_in[4];
    float* Z = (float*)d_out;

    char* ws = (char*)d_ws;
    const size_t SZ = (size_t)S_LEN * EMB * sizeof(__bf16);   // 4 MB
    __bf16* Xb = (__bf16*)(ws + 0 * SZ);
    __bf16* Q  = (__bf16*)(ws + 1 * SZ);   // [H][S][D], pre-scaled by 1/sqrt(D)
    __bf16* K  = (__bf16*)(ws + 2 * SZ);   // [H][S][D]
    __bf16* Vt = (__bf16*)(ws + 3 * SZ);   // [H][D][S]
    __bf16* Oc = (__bf16*)(ws + 4 * SZ);   // [S][H*D]

    const int nX = S_LEN * EMB;
    mha_cvt_kernel<<<nX / (256 * 8), 256, 0, stream>>>(X, Xb);

    const size_t ldsW = (size_t)HD * (EMB + 8) * sizeof(__bf16);  // ~65 KB
    mha_proj_kernel<<<NH * (S_LEN / 256), 256, ldsW, stream>>>(Xb, Wq, Wk, Wv, Q, K, Vt);

    mha_attn_kernel<<<NH * (S_LEN / 64), 128, 0, stream>>>(Q, K, Vt, Oc);

    mha_outproj_kernel<<<S_LEN / 128, 256, ldsW, stream>>>(Oc, W0, Z);
}
